// RnnEncoder_15272903704692
// MI455X (gfx1250) — matrix operand order repaired and checked by reading, not compiled, not run on hardware
//
#include <hip/hip_runtime.h>
#include <hip/hip_bf16.h>

// ---------------------------------------------------------------------------
// 3-layer LSTM (AWD-LSTM eval forward) for MI455X / gfx1250, wave32 + WMMA.
// S=512, B=64, V=50000, E=400, H=1150.
//
//   * Input projections (x_t @ WihT + bih + bhh) for ALL timesteps hoisted into
//     one large bf16 WMMA GEMM per layer -> G[S*B, Npad] (f32), 128x128 tiles.
//   * Recurrent part (h @ WhhT) runs in ONE persistent kernel per layer:
//     Whh DMA'd to LDS once via the Tensor Data Mover (tensor_load_to_lds),
//     gate-interleaved packing (n' = unit*4+gate) so each workgroup does its
//     own LSTM cell update; c state in VGPRs; ping-pong h buffer + one grid
//     barrier per timestep.
// ---------------------------------------------------------------------------

typedef __attribute__((ext_vector_type(16))) __bf16        v16bf;
typedef __attribute__((ext_vector_type(8)))  float         v8f;
typedef __attribute__((ext_vector_type(4)))  unsigned int  v4u;
typedef __attribute__((ext_vector_type(8)))  unsigned int  v8u;

union Frag16 { v16bf v; uint4 q[2]; };

// float -> bf16 (round to nearest even), stored as raw u16
__device__ inline unsigned short f2bf(float f) {
  unsigned int u = __float_as_uint(f);
  unsigned int lsb = (u >> 16) & 1u;
  u += 0x7fffu + lsb;
  return (unsigned short)(u >> 16);
}

// A fragment: 16x32 bf16, src row-major [M][ld].
// Lanes 0-15: M=lane,   K = k0+0..15
// Lanes16-31: M=lane-16,K = k0+16..31
__device__ inline v16bf load_a_frag(const unsigned short* __restrict__ base,
                                    int ld, int m0, int k0, int lane) {
  const int row = m0 + (lane & 15);
  const int kb  = k0 + ((lane & 16) ? 16 : 0);
  const uint4* p = reinterpret_cast<const uint4*>(base + (size_t)row * ld + kb);
  Frag16 f;
  f.q[0] = p[0];   // k = kb .. kb+7
  f.q[1] = p[1];   // k = kb+8 .. kb+15
  return f.v;
}

// B fragment: 32x16 bf16, src column-major [N][ldk] (ldk = padded K).
// Lanes 0-15: N=n0+lane,    K = k0+0..15
// Lanes16-31: N=n0+lane-16, K = k0+16..31
__device__ inline v16bf load_b_frag_cm(const unsigned short* __restrict__ base,
                                       int ldk, int n0, int k0, int lane) {
  const int col = n0 + (lane & 15);
  const int kb  = k0 + ((lane & 16) ? 16 : 0);
  const uint4* p = reinterpret_cast<const uint4*>(base + (size_t)col * ldk + kb);
  Frag16 f;
  f.q[0] = p[0];   // k = kb .. kb+7
  f.q[1] = p[1];   // k = kb+8 .. kb+15
  return f.v;
}

__device__ inline v8f wmma_bf16(v16bf a, v16bf b, v8f c) {
  return __builtin_amdgcn_wmma_f32_16x16x32_bf16(false, a, false, b,
                                                 (short)0, c, false, false);
}

__device__ inline float sigf(float x) { return 1.0f / (1.0f + __expf(-x)); }

// ---------------------------------------------------------------------------
// Embedding gather -> bf16 rows padded to KpadE
__global__ void embed_kernel(const int* __restrict__ tokens,
                             const float* __restrict__ emb,
                             unsigned short* __restrict__ X,
                             int E, int Kpad) {
  const int row = blockIdx.x;
  const int tok = tokens[row];
  for (int e = threadIdx.x; e < Kpad; e += blockDim.x) {
    float v = (e < E) ? emb[(size_t)tok * E + e] : 0.0f;
    X[(size_t)row * Kpad + e] = f2bf(v);
  }
}

// Pack weight [4*Hu, Din] (rows: gate*Hu+unit) -> bf16 column-major
// Wp[n' * Kpad + k], n' = unit*4 + gate, zero padded.
__global__ void pack_w_kernel(const float* __restrict__ W,
                              unsigned short* __restrict__ Wp,
                              int Hu, int Din, int Kpad) {
  const int n = blockIdx.x;
  const int unit = n >> 2, gate = n & 3;
  for (int k = threadIdx.x; k < Kpad; k += blockDim.x) {
    float v = 0.0f;
    if (unit < Hu && k < Din) v = W[(size_t)(gate * Hu + unit) * Din + k];
    Wp[(size_t)n * Kpad + k] = f2bf(v);
  }
}

__global__ void pack_bias_kernel(const float* __restrict__ bih,
                                 const float* __restrict__ bhh,
                                 float* __restrict__ bp, int Hu, int Npad) {
  const int n = blockIdx.x * blockDim.x + threadIdx.x;
  if (n >= Npad) return;
  const int unit = n >> 2, gate = n & 3;
  bp[n] = (unit < Hu) ? (bih[gate * Hu + unit] + bhh[gate * Hu + unit]) : 0.0f;
}

__global__ void init_h_kernel(const float* __restrict__ h0,
                              unsigned short* __restrict__ hbuf,
                              int Hu, int Kpad) {
  const int b = blockIdx.x;
  for (int u = threadIdx.x; u < Kpad; u += blockDim.x) {
    float v = (u < Hu) ? h0[(size_t)b * Hu + u] : 0.0f;
    hbuf[(size_t)b * Kpad + u] = f2bf(v);
  }
}

// ---------------------------------------------------------------------------
// Batched input-projection GEMM: G = X(bf16) @ Wp + bias, f32 out.
// Block = 256 threads (8 waves). Tile 128(M) x 128(N); wave (mw,nw) owns a
// 64x32 sub-tile: 8 WMMAs per 6 fragment loads, A reused x2, B reused x4.
__global__ __launch_bounds__(256)
void wmma_gemm_bias_kernel(const unsigned short* __restrict__ X,
                           const unsigned short* __restrict__ Wp,
                           const float* __restrict__ bias,
                           float* __restrict__ G,
                           int Kpad, int Npad) {
  const int lane = threadIdx.x & 31;
  const int wave = threadIdx.x >> 5;          // 0..7
  const int nw = wave & 3;                    // N sub-tile (4 x 32)
  const int mw = wave >> 2;                   // M sub-tile (2 x 64)
  const int n0 = blockIdx.x * 128 + nw * 32;  // wave's cols: n0, n0+16
  const int m0 = blockIdx.y * 128 + mw * 64;  // wave's rows: 4 blocks of 16

  v8f acc[4][2];
#pragma unroll
  for (int nb = 0; nb < 2; ++nb) {
    const float bv = bias[n0 + nb * 16 + (lane & 15)];
#pragma unroll
    for (int mb = 0; mb < 4; ++mb)
#pragma unroll
      for (int r = 0; r < 8; ++r) acc[mb][nb][r] = bv;
  }

  for (int k0 = 0; k0 < Kpad; k0 += 32) {
    // prefetch next K-chunk of this wave's weight columns (speculative)
    __builtin_prefetch(Wp + (size_t)(n0 + (lane & 15)) * Kpad + k0 + 32, 0, 1);
    const v16bf bf0 = load_b_frag_cm(Wp, Kpad, n0,      k0, lane);
    const v16bf bf1 = load_b_frag_cm(Wp, Kpad, n0 + 16, k0, lane);
#pragma unroll
    for (int mb = 0; mb < 4; ++mb) {
      const v16bf af = load_a_frag(X, Kpad, m0 + mb * 16, k0, lane);
      acc[mb][0] = wmma_bf16(af, bf0, acc[mb][0]);
      acc[mb][1] = wmma_bf16(af, bf1, acc[mb][1]);
    }
  }

#pragma unroll
  for (int nb = 0; nb < 2; ++nb) {
    const int nc = n0 + nb * 16 + (lane & 15);
#pragma unroll
    for (int mb = 0; mb < 4; ++mb) {
      const int mbase = m0 + mb * 16 + ((lane & 16) ? 8 : 0);
#pragma unroll
      for (int r = 0; r < 8; ++r)
        G[(size_t)(mbase + r) * Npad + nc] = acc[mb][nb][r];
    }
  }
}

// ---------------------------------------------------------------------------
// Device-wide sense-reversal barrier (persistent grid).
__device__ inline void grid_barrier(unsigned* cnt, unsigned* gen, unsigned nblk) {
  __threadfence();
  __syncthreads();
  if (threadIdx.x == 0) {
    volatile unsigned* vgen = gen;
    const unsigned g = *vgen;
    if (atomicAdd(cnt, 1u) == nblk - 1u) {
      *cnt = 0u;
      __threadfence();
      atomicAdd(gen, 1u);
    } else {
      while (*vgen == g) { __builtin_amdgcn_s_sleep(1); }
    }
    __threadfence();
  }
  __syncthreads();
}

// ---------------------------------------------------------------------------
// Persistent recurrent kernel. Block = 64 threads (2 waves), owns 32 packed
// gate-columns = 8 hidden units. Whh columns DMA'd into LDS once by the TDM
// and reused for all 512 steps. c-state in registers; h ping-pongs in global;
// one grid barrier per step.
__global__ __launch_bounds__(64)
void lstm_recurrent_kernel(const float* __restrict__ G,            // [S*64, Npad]
                           const unsigned short* __restrict__ Whhp,// [Npad, Kpad] cm
                           unsigned short* __restrict__ hbuf,      // [2, 64, Kpad]
                           const float* __restrict__ c0,           // [64, Hu]
                           unsigned short* __restrict__ Ybf,       // [S*64, Kpad] or null
                           float* __restrict__ Yf32,               // [S*64, Eout] or null
                           int S, int Npad, int Kpad, int Hu, int Eout,
                           unsigned* bar_cnt, unsigned* bar_gen, unsigned nblocks) {
  extern __shared__ char smem_raw[];
  unsigned short* wsh = (unsigned short*)smem_raw;                  // [32][Kpad]
  float* gbuf = (float*)(smem_raw + (size_t)32 * Kpad * sizeof(unsigned short)); // [32][64]

  const int tid  = threadIdx.x;      // 0..63
  const int lane = tid & 31;
  const int wave = tid >> 5;         // 0..1
  const int nblock0 = blockIdx.x * 32;         // global packed-col base
  const int n0loc = wave * 16;                 // wave's col base in block
  const int nglob = nblock0 + n0loc;

  // ---- One-time TDM DMA: 2D tile (Kpad contiguous x 32 cols) Global->LDS.
  // D# built per CDNA5 ISA 8.3/8.4: count=1, type=2(image), data_size=2B,
  // tensor_dim0=Kpad, tensor_dim1=Npad, tile_dim0=Kpad, tile_dim1=32,
  // tensor_dim0_stride=Kpad. global_addr points at the tile start.
  if (wave == 0) {
    const unsigned long long ga =
        (unsigned long long)(uintptr_t)(Whhp + (size_t)nblock0 * Kpad);
    const unsigned ldsoff = (unsigned)(uintptr_t)wsh;  // flat LDS addr low 32b
    v4u g0;
    g0[0] = 1u;                                        // count=1, user desc
    g0[1] = ldsoff;                                    // lds_addr (bytes)
    g0[2] = (unsigned)ga;                              // global_addr[31:0]
    g0[3] = (unsigned)((ga >> 32) & 0x01ffffffu) | (2u << 30); // [56:32]|type=2
    v8u g1;
    g1[0] = 1u << 16;                                  // data_size = 2 bytes
    g1[1] = ((unsigned)Kpad & 0xffffu) << 16;          // tensor_dim0[15:0]
    g1[2] = (((unsigned)Npad & 0xffffu) << 16) | ((unsigned)Kpad >> 16);
    g1[3] = (((unsigned)Kpad & 0xffffu) << 16) | ((unsigned)Npad >> 16);
    g1[4] = 32u;                                       // tile_dim1 = 32
    g1[5] = (unsigned)Kpad;                            // dim0_stride[31:0]
    g1[6] = 0u;
    g1[7] = 0u;
    asm volatile("tensor_load_to_lds %0, %1"
                 :: "s"(g0), "s"(g1)
                 : "memory");
    __builtin_amdgcn_s_wait_tensorcnt((short)0);
  }

  // Cell-state registers: thread -> unit ul = tid/8 (0..7), batches bbase..+7
  const int ul = tid >> 3;
  const int bbase = (tid & 7) * 8;
  const int ug = blockIdx.x * 8 + ul;          // global hidden-unit index
  float creg[8];
#pragma unroll
  for (int j = 0; j < 8; ++j)
    creg[j] = (ug < Hu) ? c0[(size_t)(bbase + j) * Hu + ug] : 0.0f;

  __syncthreads();   // LDS weights visible to both waves

  for (int t = 0; t < S; ++t) {
    const unsigned short* hsrc = hbuf + (size_t)(t & 1) * 64 * Kpad;
    unsigned short* hdst = hbuf + (size_t)((t + 1) & 1) * 64 * Kpad;
    const float* Gt = G + (size_t)t * 64 * Npad;

    // Prefetch next step's G rows at our column window (one line per row).
    if (t + 1 < S)
      __builtin_prefetch(Gt + (size_t)64 * Npad + (size_t)tid * Npad + nblock0, 0, 1);

    // ---- gates(64 x 32cols) = G[t] + h @ WhhT ----
    v8f acc[4];
    {
      const int nc = nglob + (lane & 15);
#pragma unroll
      for (int mb = 0; mb < 4; ++mb) {
        const int mbase = mb * 16 + ((lane & 16) ? 8 : 0);
#pragma unroll
        for (int r = 0; r < 8; ++r)
          acc[mb][r] = Gt[(size_t)(mbase + r) * Npad + nc];
      }
    }
    for (int k0 = 0; k0 < Kpad; k0 += 32) {
      const v16bf bf = load_b_frag_cm(wsh, Kpad, n0loc, k0, lane);
#pragma unroll
      for (int mb = 0; mb < 4; ++mb) {
        const v16bf af = load_a_frag(hsrc, Kpad, mb * 16, k0, lane);
        acc[mb] = wmma_bf16(af, bf, acc[mb]);
      }
    }

    // Spill gate tile to LDS as [col_local][batch]
    {
      const int cl = n0loc + (lane & 15);
#pragma unroll
      for (int mb = 0; mb < 4; ++mb) {
        const int mbase = mb * 16 + ((lane & 16) ? 8 : 0);
#pragma unroll
        for (int r = 0; r < 8; ++r)
          gbuf[cl * 64 + (mbase + r)] = acc[mb][r];
      }
    }
    __syncthreads();

    // ---- LSTM cell update (PyTorch gate order i,f,g,o interleaved) ----
#pragma unroll
    for (int j = 0; j < 8; ++j) {
      const int b = bbase + j;
      const float gi = gbuf[(ul * 4 + 0) * 64 + b];
      const float gf = gbuf[(ul * 4 + 1) * 64 + b];
      const float gg = gbuf[(ul * 4 + 2) * 64 + b];
      const float go = gbuf[(ul * 4 + 3) * 64 + b];
      const float c = sigf(gf) * creg[j] + sigf(gi) * tanhf(gg);
      creg[j] = c;
      const float h = sigf(go) * tanhf(c);
      hdst[(size_t)b * Kpad + ug] = f2bf(h);
      if (Ybf)  Ybf[((size_t)t * 64 + b) * Kpad + ug] = f2bf(h);
      if (Yf32 && ug < Eout) Yf32[((size_t)t * 64 + b) * Eout + ug] = h;
    }
    __syncthreads();  // protect gbuf WAR before next step's writes

    grid_barrier(bar_cnt, bar_gen, nblocks); // publish hdst to all blocks
  }
}

// ---------------------------------------------------------------------------
extern "C" void kernel_launch(void* const* d_in, const int* in_sizes, int n_in,
                              void* d_out, int out_size, void* d_ws, size_t ws_size,
                              hipStream_t stream) {
  (void)in_sizes; (void)n_in; (void)out_size; (void)ws_size;
  const int S = 512, B = 64, E = 400, H = 1150;
  const int KpadE = 416, KpadH = 1152;           // mult of 32
  const int NpadH = 4608, NpadE = 1664;          // 4 * padded units
  const int M = S * B;                           // 32768 rows

  const int*   tokens = (const int*)d_in[0];
  const float* emb    = (const float*)d_in[1];
  const float* Wih[3] = {(const float*)d_in[2],  (const float*)d_in[8],  (const float*)d_in[14]};
  const float* Whh[3] = {(const float*)d_in[3],  (const float*)d_in[9],  (const float*)d_in[15]};
  const float* bih[3] = {(const float*)d_in[4],  (const float*)d_in[10], (const float*)d_in[16]};
  const float* bhh[3] = {(const float*)d_in[5],  (const float*)d_in[11], (const float*)d_in[17]};
  const float* h0[3]  = {(const float*)d_in[6],  (const float*)d_in[12], (const float*)d_in[18]};
  const float* c0[3]  = {(const float*)d_in[7],  (const float*)d_in[13], (const float*)d_in[19]};

  // ---- workspace carve ----
  char* ws = (char*)d_ws;
  size_t off = 0;
  auto alloc = [&](size_t bytes) -> void* {
    void* p = ws + off;
    off = (off + bytes + 255) & ~(size_t)255;
    return p;
  };
  unsigned*       bar  = (unsigned*)alloc(256);
  unsigned short* Xbf  = (unsigned short*)alloc((size_t)M * KpadE * 2);
  unsigned short* Y0   = (unsigned short*)alloc((size_t)M * KpadH * 2);
  unsigned short* Y1   = (unsigned short*)alloc((size_t)M * KpadH * 2);
  float*          G    = (float*)alloc((size_t)M * NpadH * 4);
  unsigned short* Wihp[3] = {(unsigned short*)alloc((size_t)NpadH * KpadE * 2),
                             (unsigned short*)alloc((size_t)NpadH * KpadH * 2),
                             (unsigned short*)alloc((size_t)NpadE * KpadH * 2)};
  unsigned short* Whhp[3] = {(unsigned short*)alloc((size_t)NpadH * KpadH * 2),
                             (unsigned short*)alloc((size_t)NpadH * KpadH * 2),
                             (unsigned short*)alloc((size_t)NpadE * KpadE * 2)};
  float* biasp[3] = {(float*)alloc(NpadH * 4), (float*)alloc(NpadH * 4), (float*)alloc(NpadE * 4)};
  unsigned short* hbuf = (unsigned short*)alloc((size_t)2 * 64 * KpadH * 2);

  hipMemsetAsync(bar, 0, 2 * sizeof(unsigned), stream);

  const int Kin[3]  = {KpadE, KpadH, KpadH};   // padded input K per layer
  const int Din[3]  = {E, H, H};               // real input dim per layer
  const int Krec[3] = {KpadH, KpadH, KpadE};   // padded recurrent K per layer
  const int Npad[3] = {NpadH, NpadH, NpadE};
  const int Hu[3]   = {H, H, E};               // real hidden size per layer

  // Embedding -> bf16
  embed_kernel<<<M, 128, 0, stream>>>(tokens, emb, Xbf, E, KpadE);

  // Weight / bias packing (gate-interleaved, bf16, column-major)
  for (int l = 0; l < 3; ++l) {
    pack_w_kernel<<<Npad[l], 128, 0, stream>>>(Wih[l], Wihp[l], Hu[l], Din[l], Kin[l]);
    pack_w_kernel<<<Npad[l], 128, 0, stream>>>(Whh[l], Whhp[l], Hu[l], Hu[l], Krec[l]);
    pack_bias_kernel<<<(Npad[l] + 127) / 128, 128, 0, stream>>>(bih[l], bhh[l], biasp[l], Hu[l], Npad[l]);
  }

  const unsigned short* Xin[3]  = {Xbf, Y0, Y1};
  unsigned short*       Yout[3] = {Y0, Y1, nullptr};

  for (int l = 0; l < 3; ++l) {
    // Batched input projection for all timesteps: G = X @ WihT + (bih+bhh)
    dim3 g(Npad[l] / 128, M / 128);
    wmma_gemm_bias_kernel<<<g, 256, 0, stream>>>(Xin[l], Wihp[l], biasp[l], G, Kin[l], Npad[l]);

    // Recurrence
    init_h_kernel<<<B, 128, 0, stream>>>(h0[l], hbuf, Hu[l], Krec[l]);
    const int nblocks = Npad[l] / 32;
    const size_t shmem = (size_t)32 * Krec[l] * 2 + (size_t)32 * 64 * 4;
    lstm_recurrent_kernel<<<nblocks, 64, shmem, stream>>>(
        G, Whhp[l], hbuf, c0[l], Yout[l], (l == 2) ? (float*)d_out : nullptr,
        S, Npad[l], Krec[l], Hu[l], E, bar, bar + 1, (unsigned)nblocks);
  }
}
